// StochasticStateModel_19945828123156
// MI455X (gfx1250) — matrix operand order repaired and checked by reading, not compiled, not run on hardware
//
#include <hip/hip_runtime.h>

#define ZLEV   34
#define NPIX   (512 * 1024)
#define ZPAD   48
#define NPAIRS (NPIX / 32)   // 16384 pairs of 16-cell groups (M=32 per wave)

typedef __attribute__((ext_vector_type(16))) __bf16 v16bf;
typedef __attribute__((ext_vector_type(8)))  float  v8f;

__device__ __forceinline__ unsigned bf16b(float f) {
  unsigned u = __builtin_bit_cast(unsigned, f);
  return ((u + 0x7FFFu + ((u >> 16) & 1u)) >> 16) & 0xFFFFu;  // RNE f32->bf16
}

// Feature k of cell n. k: [0,15)=qt_pred, [15,33)=sli_pred, 33=sst,
// [34,68)=qt, [68,102)=sli, 102=solin, 103=bias(1.0), >=104 zero pad.
__device__ __forceinline__ float load_feat(
    int k, int n,
    const float* __restrict__ qt_pred, const float* __restrict__ sli_pred,
    const float* __restrict__ sst,     const float* __restrict__ qt,
    const float* __restrict__ sli,     const float* __restrict__ solin) {
  if (k < 15)   return qt_pred[(size_t)k * NPIX + n];
  if (k < 33)   return sli_pred[(size_t)(k - 15) * NPIX + n];
  if (k == 33)  return sst[n];
  if (k < 68)   return qt[(size_t)(k - 34) * NPIX + n];
  if (k < 102)  return sli[(size_t)(k - 68) * NPIX + n];
  if (k == 102) return solin[n];
  if (k == 103) return 1.0f;
  return 0.0f;
}

extern "C" __global__ __launch_bounds__(512)
void residual_eta_gemm(
    const float* __restrict__ qt_pred, const float* __restrict__ sli_pred,
    const float* __restrict__ sst,     const float* __restrict__ qt,
    const float* __restrict__ sli,     const float* __restrict__ solin,
    const float* __restrict__ w_qt,    const float* __restrict__ b_qt,
    const float* __restrict__ w_sli,   const float* __restrict__ b_sli,
    const int*   __restrict__ eta,     float* __restrict__ out)
{
  extern __shared__ char smem[];
  unsigned short* wlds = (unsigned short*)smem;   // 192 KB: 192 B-tiles x 1 KB
  float* tbuf = (float*)(smem + 192 * 1024);      // 16 waves x (48 x 16) f32

  const int tid = threadIdx.x;

  // ---- Stage all weights (2 models x 8 eta), bf16, pre-swizzled as WMMA ----
  // B-operand tiles: tile = ((model*8+e)*4+c)*3+zt ; within a tile, lane l
  // holds 16 bf16 = W[e][c*32 + (l<16?0:16) + j][zt*16 + (l&15)], j=0..15.
  for (int t = tid; t < 98304; t += 512) {
    int j     = t & 15;
    int lane  = (t >> 4) & 31;
    int tileI = t >> 9;              // [0,192)
    int zt    = tileI % 3;
    int q     = tileI / 3;
    int c     = q & 3;
    int e     = (q >> 2) & 7;
    int model = q >> 5;
    int k = c * 32 + ((lane < 16) ? 0 : 16) + j;
    int z = zt * 16 + (lane & 15);
    const float* w = model ? w_sli : w_qt;   // [8, 103, 34]
    const float* b = model ? b_sli : b_qt;   // [8, 34]
    float v = 0.0f;
    if (z < ZLEV) {
      if (k < 103)       v = w[(e * 103 + k) * ZLEV + z];
      else if (k == 103) v = b[e * ZLEV + z];   // bias row, pairs with A==1.0
    }
    wlds[t] = (unsigned short)bf16b(v);
  }
  __syncthreads();

  const int lane = tid & 31;
  const int wave = tid >> 5;
  const int row  = lane & 15;          // A-matrix row (cell within group)
  const int hi   = (lane >> 4) & 1;    // lane half
  float* buf = tbuf + wave * (ZPAD * 16);

  const int gw = blockIdx.x * 16 + wave;
  const int nw = gridDim.x * 16;       // 4096 waves -> exactly 4 pairs each

  for (int g = gw; g < NPAIRS; g += nw) {
    const int n0 = g << 5;             // 32 cells per wave iteration
    const int nA = n0 + row;           // group 0 rows
    const int nB = n0 + 16 + row;      // group 1 rows
    const int etaA = eta[nA];
    const int etaB = eta[nB];

    v8f acc[2][2][3];                  // [group][model][ztile]
    #pragma unroll
    for (int grp = 0; grp < 2; ++grp)
      #pragma unroll
      for (int m = 0; m < 2; ++m)
        #pragma unroll
        for (int zt = 0; zt < 3; ++zt) acc[grp][m][zt] = (v8f)(0.0f);

    #pragma unroll
    for (int c = 0; c < 4; ++c) {
      // A-operand (bf16 16x32 layout): lanes<16 hold K = c*32+{0..15},
      // lanes>=16 hold K = c*32+{16..31}; VGPR p packs (k, k+1).
      const int kbase = c * 32 + (hi ? 16 : 0);
      union { unsigned u[8]; v16bf v; } A0, A1;
      #pragma unroll
      for (int p = 0; p < 8; ++p) {
        const int i0 = 2 * p;
        const int k0 = kbase + i0;
        float a0 = load_feat(k0,     nA, qt_pred, sli_pred, sst, qt, sli, solin);
        float a1 = load_feat(k0 + 1, nA, qt_pred, sli_pred, sst, qt, sli, solin);
        float b0 = load_feat(k0,     nB, qt_pred, sli_pred, sst, qt, sli, solin);
        float b1 = load_feat(k0 + 1, nB, qt_pred, sli_pred, sst, qt, sli, solin);
        A0.u[p] = bf16b(a0) | (bf16b(a1) << 16);
        A1.u[p] = bf16b(b0) | (bf16b(b1) << 16);
      }
      // Masked eta accumulation: rows whose eta != e contribute zero.
      for (int e = 0; e < 8; ++e) {
        union { unsigned u[8]; v16bf v; } Am0, Am1;
        #pragma unroll
        for (int p = 0; p < 8; ++p) {
          Am0.u[p] = (etaA == e) ? A0.u[p] : 0u;
          Am1.u[p] = (etaB == e) ? A1.u[p] : 0u;
        }
        #pragma unroll
        for (int m = 0; m < 2; ++m) {
          #pragma unroll
          for (int zt = 0; zt < 3; ++zt) {
            const int tile = ((m * 8 + e) * 4 + c) * 3 + zt;
            const v16bf* Bp = (const v16bf*)(smem + tile * 1024 + lane * 32);
            v16bf B = *Bp;             // one 1KB tile feeds two WMMAs
            acc[0][m][zt] = __builtin_amdgcn_wmma_f32_16x16x32_bf16(
                false, Am0.v, false, B, (short)0, acc[0][m][zt], false, false);
            acc[1][m][zt] = __builtin_amdgcn_wmma_f32_16x16x32_bf16(
                false, Am1.v, false, B, (short)0, acc[1][m][zt], false, false);
          }
        }
      }
    }

    // ---- Transpose C tiles through LDS for coalesced residual add/store ----
    // C layout: VGPR r -> cell = r + hi*8 ; z = zt*16 + (lane&15).
    #pragma unroll
    for (int grp = 0; grp < 2; ++grp) {
      const int base = n0 + grp * 16;
      #pragma unroll
      for (int m = 0; m < 2; ++m) {
        const float* pred = m ? sli_pred : qt_pred;
        float* op = out + (size_t)m * ZLEV * NPIX;
        #pragma unroll
        for (int zt = 0; zt < 3; ++zt)
          #pragma unroll
          for (int r = 0; r < 8; ++r)
            buf[(zt * 16 + row) * 16 + r + hi * 8] = acc[grp][m][zt][r];
        // wave-private buffer; LDS ops from one wave are in order
        for (int it = 0; it < 17; ++it) {
          int z    = 2 * it + hi;        // covers z = 0..33
          int cell = row;
          float y  = buf[z * 16 + cell];
          size_t idx = (size_t)z * NPIX + (size_t)(base + cell);
          op[idx] = pred[idx] + y;       // DT_RATIO == 1.0
        }
      }
    }
  }
}

extern "C" void kernel_launch(void* const* d_in, const int* in_sizes, int n_in,
                              void* d_out, int out_size, void* d_ws, size_t ws_size,
                              hipStream_t stream) {
  (void)in_sizes; (void)n_in; (void)out_size; (void)d_ws; (void)ws_size;
  const float* qt_pred  = (const float*)d_in[0];
  const float* sli_pred = (const float*)d_in[1];
  const float* sst      = (const float*)d_in[2];
  const float* qtv      = (const float*)d_in[3];
  const float* sliv     = (const float*)d_in[4];
  const float* solin    = (const float*)d_in[5];
  const float* w_qt     = (const float*)d_in[6];
  const float* b_qt     = (const float*)d_in[7];
  const float* w_sli    = (const float*)d_in[8];
  const float* b_sli    = (const float*)d_in[9];
  const int*   eta      = (const int*)d_in[10];
  float* out = (float*)d_out;

  const size_t shmem = 192 * 1024 + (size_t)16 * ZPAD * 16 * sizeof(float); // 240 KB
  (void)hipFuncSetAttribute((const void*)residual_eta_gemm,
                            hipFuncAttributeMaxDynamicSharedMemorySize,
                            (int)shmem);

  // 256 blocks x 16 waves = 4096 waves; 16384 pairs -> exactly 4 per wave.
  residual_eta_gemm<<<dim3(256), dim3(512), shmem, stream>>>(
      qt_pred, sli_pred, sst, qtv, sliv, solin,
      w_qt, b_qt, w_sli, b_sli, eta, out);
}
